// HierarchicalDynamicSlotsSoftMoE_27255862460960
// MI455X (gfx1250) — matrix operand order repaired and checked by reading, not compiled, not run on hardware
//
#include <hip/hip_runtime.h>

// ---------------------------------------------------------------------------
// MI455X (gfx1250) implementation of HierarchicalDynamicSlotsSoftMoE.
// Heavy GEMMs: v_wmma_f32_16x16x32_bf16 (wave32 WMMA), bf16 in, f32 accum.
// GEMM tile staging uses CDNA5 GLOBAL_LOAD_ASYNC_TO_LDS_B128 (ASYNCcnt) with
// double-buffered LDS tiles so the async DMA for tile k+1 overlaps the WMMAs
// of tile k. Weights are converted fp32->bf16 *and transposed* once per call
// so every weight GEMM stages with pure async b128 copies.
// Small K=16/64 contractions stay on VALU (bandwidth-trivial, ~2 GF total).
// ---------------------------------------------------------------------------

typedef __bf16 bf16;
typedef __attribute__((ext_vector_type(16))) __bf16 v16bf;
typedef __attribute__((ext_vector_type(8)))  __bf16 v8bf;
typedef __attribute__((ext_vector_type(4)))  __bf16 v4bf;
typedef __attribute__((ext_vector_type(8)))  float  v8f;
typedef long long i64;

__device__ __forceinline__ v8f wmma_bf16(v16bf a, v16bf b, v8f c) {
  // (neg_a, A, neg_b, B, c_mod, C, reuse_a, reuse_b)
  return __builtin_amdgcn_wmma_f32_16x16x32_bf16(false, a, false, b, (short)0, c,
                                                 false, false);
}

// Async global->LDS 16-byte copy (per lane). LDS byte offset is the low 32
// bits of the generic pointer for __shared__ objects on gfx1250.
__device__ __forceinline__ void async_cp16(void* lds, const void* g) {
  unsigned l = (unsigned)(unsigned long long)lds;
  asm volatile("global_load_async_to_lds_b128 %0, %1, off"
               :: "v"(l), "v"(g) : "memory");
}
__device__ __forceinline__ void async_wait0() {
  asm volatile("s_wait_asynccnt 0" ::: "memory");
}

// ----------------------------- reductions ---------------------------------
__device__ __forceinline__ float blk_sum256(float v, volatile float* red) {
#pragma unroll
  for (int o = 16; o > 0; o >>= 1) v += __shfl_down(v, o, 32);
  int lane = threadIdx.x & 31, wv = threadIdx.x >> 5;
  if (lane == 0) red[wv] = v;
  __syncthreads();
  if (threadIdx.x == 0) {
    float t = 0.f;
    for (int i = 0; i < 8; ++i) t += red[i];
    red[0] = t;
  }
  __syncthreads();
  float r = red[0];
  __syncthreads();
  return r;
}

__device__ __forceinline__ float blk_max256(float v, volatile float* red) {
#pragma unroll
  for (int o = 16; o > 0; o >>= 1) v = fmaxf(v, __shfl_down(v, o, 32));
  int lane = threadIdx.x & 31, wv = threadIdx.x >> 5;
  if (lane == 0) red[wv] = v;
  __syncthreads();
  if (threadIdx.x == 0) {
    float t = red[0];
    for (int i = 1; i < 8; ++i) t = fmaxf(t, red[i]);
    red[0] = t;
  }
  __syncthreads();
  float r = red[0];
  __syncthreads();
  return r;
}

// ------------------------- elementwise kernels ----------------------------
// Tiled convert + transpose: in (K,N) f32 -> out (N,K) bf16. Batched over z.
// grid: (N/32, K/32, batch), block 256.
__global__ __launch_bounds__(256) void k_cvt_t_bf16(const float* __restrict__ in,
                                                    bf16* __restrict__ out,
                                                    int Kd, int Nd) {
  __shared__ float tile[32][33];
  int z = blockIdx.z;
  const float* ib = in + (i64)z * Kd * Nd;
  bf16* ob = out + (i64)z * Kd * Nd;
  int k0 = blockIdx.y * 32;
  int n0 = blockIdx.x * 32;
  int tx = threadIdx.x & 31;
  int ty = threadIdx.x >> 5;  // 0..7
#pragma unroll
  for (int i = 0; i < 4; ++i)
    tile[ty + i * 8][tx] = ib[(i64)(k0 + ty + i * 8) * Nd + n0 + tx];
  __syncthreads();
#pragma unroll
  for (int i = 0; i < 4; ++i)
    ob[(i64)(n0 + ty + i * 8) * Kd + k0 + tx] = (bf16)tile[tx][ty + i * 8];
}

// rmsnorm: row -> x / max(||x||,eps) * sqrt(d) * gamma
// mode 0: in_row == out_row.  mode 1: out row r=(b*e2+e)*n2+n, in row (b*n2+n)*e2+e
__global__ __launch_bounds__(256) void k_rmsnorm(const float* __restrict__ in,
                                                 const float* __restrict__ gamma,
                                                 float* __restrict__ outf,
                                                 bf16* __restrict__ outb,
                                                 int d, int mode, int n2, int e2) {
  __shared__ float red[8];
  i64 row = blockIdx.x;
  i64 in_row = row;
  if (mode == 1) {
    int per_b = e2 * n2;
    int b = (int)(row / per_b), rem = (int)(row % per_b);
    int e = rem / n2, n = rem % n2;
    in_row = ((i64)b * n2 + n) * e2 + e;
  }
  const float* x = in + in_row * d;
  float ss = 0.f;
  for (int c = threadIdx.x; c < d; c += 256) {
    float v = x[c];
    ss += v * v;
  }
  float tot = blk_sum256(ss, red);
  float s = sqrtf((float)d) / fmaxf(sqrtf(tot), 1e-8f);
  for (int c = threadIdx.x; c < d; c += 256) {
    float v = x[c] * s * gamma[c];
    if (outf) outf[row * d + c] = v;
    if (outb) outb[row * d + c] = (bf16)v;
  }
}

// out[r][c] = mean_{g<grp} in[r*grp+g][c]
__global__ __launch_bounds__(256) void k_group_mean(const float* __restrict__ in,
                                                    float* __restrict__ outf,
                                                    bf16* __restrict__ outb,
                                                    int d, int grp, i64 total) {
  i64 idx = (i64)blockIdx.x * blockDim.x + threadIdx.x;
  if (idx >= total) return;
  i64 r = idx / d;
  int c = (int)(idx % d);
  const float* p = in + (r * grp) * (i64)d + c;
  float s = 0.f;
  for (int g = 0; g < grp; ++g) s += p[(i64)g * d];
  s *= (1.0f / (float)grp);
  if (outf) outf[idx] = s;
  if (outb) outb[idx] = (bf16)s;
}

__global__ __launch_bounds__(256) void k_gates(const float* __restrict__ gm,
                                               const float* __restrict__ Wg,
                                               const float* __restrict__ bg0,
                                               float* __restrict__ gates, int d) {
  __shared__ float red[8];
  i64 row = blockIdx.x;
  const float* x = gm + row * d;
  float s = 0.f;
  for (int c = threadIdx.x; c < d; c += 256) s += x[c] * Wg[c];
  float tot = blk_sum256(s, red);
  if (threadIdx.x == 0) {
    float z = tot + bg0[0];
    float sg = 1.f / (1.f + expf(-z));
    gates[row] = fminf(fmaxf(sg, 0.01f), 1.0f);
  }
}

// ------------------------------ softmaxes ---------------------------------
__global__ __launch_bounds__(256) void k_softmax_row(const float* __restrict__ in,
                                                     float* __restrict__ outf,
                                                     bf16* __restrict__ outb, int C) {
  __shared__ float red[8];
  i64 row = blockIdx.x;
  const float* x = in + row * (i64)C;
  float mx = -3.402823466e38f;
  for (int c = threadIdx.x; c < C; c += 256) mx = fmaxf(mx, x[c]);
  mx = blk_max256(mx, red);
  float s = 0.f;
  for (int c = threadIdx.x; c < C; c += 256) s += expf(x[c] - mx);
  s = blk_sum256(s, red);
  float inv = 1.0f / s;
  for (int c = threadIdx.x; c < C; c += 256) {
    float v = expf(x[c] - mx) * inv;
    if (outf) outf[row * (i64)C + c] = v;
    if (outb) outb[row * (i64)C + c] = (bf16)v;
  }
}

// softmax over rows (axis 0) of an RxC matrix, write transposed bf16 (CxR).
// grid: (C/256, batch)
__global__ __launch_bounds__(256) void k_colsoftmax_T(const float* __restrict__ in,
                                                      bf16* __restrict__ outT,
                                                      int R, int C) {
  int z = blockIdx.y;
  int c = blockIdx.x * 256 + threadIdx.x;
  const float* p = in + (i64)z * R * C + c;
  float mx = -3.402823466e38f;
  for (int r = 0; r < R; ++r) mx = fmaxf(mx, p[(i64)r * C]);
  float s = 0.f;
  for (int r = 0; r < R; ++r) s += expf(p[(i64)r * C] - mx);
  float inv = 1.0f / s;
  bf16* o = outT + (i64)z * R * C + (i64)c * R;
  for (int r = 0; r < R; ++r) o[r] = (bf16)(expf(p[(i64)r * C] - mx) * inv);
}

// --------------------------- main WMMA GEMM -------------------------------
// C[z] = A[z] * B[bidx] (+bias[bidx])  ; A:(M,K) bf16 row-major
// BT=1: B is (N,K) row-major -> async b128 copy to LDS
// BT=0: B is (K,N) row-major -> transposed while staging (register-pipelined)
// Output: Cf (f32) or Cb (bf16), optional exact GELU.
// Requires M%128==0, N%128==0, K%32==0. 256 threads = 8 waves in 2x4 grid,
// each wave owns a 64x32 tile = 4x2 WMMA 16x16 accumulators.
// Double-buffered LDS: async copies for tile s+1 are issued before the WMMAs
// of tile s; each wave then waits its own ASYNCcnt and hits the barrier.
template <int BT>
__global__ __launch_bounds__(256) void k_gemm(
    const bf16* __restrict__ A, i64 lda, i64 sA,
    const bf16* __restrict__ B, i64 ldb, i64 sB, int bMod,
    float* __restrict__ Cf, bf16* __restrict__ Cb, i64 ldc, i64 sC,
    const float* __restrict__ bias, i64 sBias, int K, int gelu) {
  __shared__ __align__(16) bf16 As[2][128][40];  // +8 pad: bank spread, 16B rows
  __shared__ __align__(16) bf16 Bs[2][128][40];  // stored [n][k]

  const int z = blockIdx.z;
  const int bidx = (bMod > 0) ? (z % bMod) : z;
  const bf16* Ab = A + (i64)z * sA;
  const bf16* Bb = B + (i64)bidx * sB;
  const float* bp = bias ? (bias + (i64)bidx * sBias) : nullptr;
  const i64 cBase = (i64)z * sC;
  const int n0 = blockIdx.x * 128;
  const int m0 = blockIdx.y * 128;
  const int tid = threadIdx.x;
  const int wave = tid >> 5, lane = tid & 31;
  const int wm = (wave >> 2) * 64;  // wave row origin (2 waves along M)
  const int wn = (wave & 3) * 32;   // wave col origin (4 waves along N)
  const int hf = lane >> 4, lr = lane & 15;

  v8f acc[4][2];
  const v8f vz = {0.f, 0.f, 0.f, 0.f, 0.f, 0.f, 0.f, 0.f};
#pragma unroll
  for (int i = 0; i < 4; ++i)
#pragma unroll
    for (int j = 0; j < 2; ++j) acc[i][j] = vz;

  const int sr = tid >> 2;        // staging row 0..63 (+64 on 2nd sweep)
  const int sc = (tid & 3) * 8;   // staging col {0,8,16,24}

  v4bf tb[4];  // register pipeline for BT=0 B-tile transpose

  auto stageA = [&](int k0, int buf) {
#pragma unroll
    for (int it = 0; it < 2; ++it) {
      int r = it * 64 + sr;
      async_cp16(&As[buf][r][sc], &Ab[(i64)(m0 + r) * lda + k0 + sc]);
    }
  };
  auto stageB = [&](int k0, int buf) {
#pragma unroll
    for (int it = 0; it < 2; ++it) {
      int r = it * 64 + sr;
      async_cp16(&Bs[buf][r][sc], &Bb[(i64)(n0 + r) * ldb + k0 + sc]);
    }
  };
  auto loadBT = [&](int k0) {  // B (K,N): gather 4x v4bf into registers
#pragma unroll
    for (int it = 0; it < 4; ++it) {
      int idx = it * 256 + tid;       // 1024 chunks of 4
      int kk = idx >> 5;              // 0..31
      int nc = (idx & 31) * 4;        // 0..124
      tb[it] = *(const v4bf*)&Bb[(i64)(k0 + kk) * ldb + n0 + nc];
    }
  };
  auto storeBT = [&](int buf) {  // scatter registers transposed into LDS
#pragma unroll
    for (int it = 0; it < 4; ++it) {
      int idx = it * 256 + tid;
      int kk = idx >> 5;
      int nc = (idx & 31) * 4;
      Bs[buf][nc + 0][kk] = tb[it].x;
      Bs[buf][nc + 1][kk] = tb[it].y;
      Bs[buf][nc + 2][kk] = tb[it].z;
      Bs[buf][nc + 3][kk] = tb[it].w;
    }
  };

  // prologue: stage tile 0 into buffer 0
  stageA(0, 0);
  if (BT) stageB(0, 0);
  else { loadBT(0); storeBT(0); }
  async_wait0();
  __syncthreads();

  const int nsteps = K >> 5;
  for (int s = 0; s < nsteps; ++s) {
    const int cur = s & 1, nxt = cur ^ 1;
    const bool pre = (s + 1 < nsteps);
    if (pre) {
      stageA((s + 1) * 32, nxt);
      if (BT) stageB((s + 1) * 32, nxt);
      else loadBT((s + 1) * 32);
    }

    // A frag (16x32): lanes 0-15 row=lr K0..7,K16..23; lanes 16-31 K8..15,K24..31
    v16bf af[4];
#pragma unroll
    for (int i = 0; i < 4; ++i) {
      union { v16bf v; v8bf h[2]; } u;
      const bf16* ap = &As[cur][wm + i * 16 + lr][hf * 8];
      u.h[0] = *(const v8bf*)ap;
      u.h[1] = *(const v8bf*)(ap + 16);
      af[i] = u.v;
    }
    // B frag (32x16): lanes 0-15 col=lr K0..7,K16..23; lanes 16-31 K8..15,K24..31
    v16bf bfr[2];
#pragma unroll
    for (int j = 0; j < 2; ++j) {
      union { v16bf v; v8bf h[2]; } u;
      const bf16* bpw = &Bs[cur][wn + j * 16 + lr][hf * 8];
      u.h[0] = *(const v8bf*)bpw;
      u.h[1] = *(const v8bf*)(bpw + 16);
      bfr[j] = u.v;
    }
#pragma unroll
    for (int i = 0; i < 4; ++i)
#pragma unroll
      for (int j = 0; j < 2; ++j) acc[i][j] = wmma_bf16(af[i], bfr[j], acc[i][j]);

    if (!BT && pre) storeBT(nxt);  // LDS scatter after the math
    async_wait0();
    __syncthreads();
  }

  // epilogue: C frag layout: VGPR v -> row v (+8 for lanes>=16), col = lr
#pragma unroll
  for (int i = 0; i < 4; ++i)
#pragma unroll
    for (int j = 0; j < 2; ++j) {
      int n = n0 + wn + j * 16 + lr;
      float bv = bp ? bp[n] : 0.0f;
      int mb = m0 + wm + i * 16 + hf * 8;
#pragma unroll
      for (int v = 0; v < 8; ++v) {
        float val = acc[i][j][v] + bv;
        if (gelu) val = 0.5f * val * (1.0f + erff(val * 0.70710678118654752f));
        i64 off = cBase + (i64)(mb + v) * ldc + n;
        if (Cf) Cf[off] = val; else Cb[off] = (bf16)val;
      }
    }
}

// one wave per 16x16 tile, N fixed at 16, B already (N,K): for stage-1 logits
__global__ __launch_bounds__(32) void k_gemm16(
    const bf16* __restrict__ A, i64 lda, i64 sA,
    const bf16* __restrict__ Bt, i64 ldb, i64 sB,
    float* __restrict__ C, i64 ldc, i64 sC, int K) {
  const int z = blockIdx.z;
  const bf16* a = A + (i64)z * sA + (i64)blockIdx.x * 16 * lda;
  const bf16* bt = Bt + (i64)z * sB;
  const int lane = threadIdx.x;
  const int hf = lane >> 4, lr = lane & 15;
  v8f acc = {0.f, 0.f, 0.f, 0.f, 0.f, 0.f, 0.f, 0.f};
  for (int k = 0; k < K; k += 32) {
    union { v16bf v; v8bf h[2]; } ua, ub;
    const bf16* ap = &a[(i64)lr * lda + k + hf * 8];
    ua.h[0] = *(const v8bf*)ap;
    ua.h[1] = *(const v8bf*)(ap + 16);
    const bf16* bp = &bt[(i64)lr * ldb + k + hf * 8];
    ub.h[0] = *(const v8bf*)bp;
    ub.h[1] = *(const v8bf*)(bp + 16);
    acc = wmma_bf16(ua.v, ub.v, acc);
  }
  float* c = C + (i64)z * sC + (i64)blockIdx.x * 16 * ldc;
#pragma unroll
  for (int v = 0; v < 8; ++v) c[(i64)(v + hf * 8) * ldc + lr] = acc[v];
}

// --------------------- small VALU contractions (K<=64) --------------------
// slots[m][c] = sum_k combine[k][m] * x[k][c]; write bf16 in per-expert layout
// out[e][(bg*2+s)][c], e=m>>1, s=m&1.  one block per group bg (512 blocks).
__global__ __launch_bounds__(256) void k_slots(const float* __restrict__ x,
                                               const float* __restrict__ comb,
                                               bf16* __restrict__ out) {
  int bg = blockIdx.x;
  __shared__ float cs[64 * 16];
  for (int i = threadIdx.x; i < 1024; i += 256) cs[i] = comb[(i64)bg * 1024 + i];
  __syncthreads();
  float acc[16][4];
#pragma unroll
  for (int m = 0; m < 16; ++m)
#pragma unroll
    for (int j = 0; j < 4; ++j) acc[m][j] = 0.f;
  for (int k = 0; k < 64; ++k) {
    float v[4];
#pragma unroll
    for (int j = 0; j < 4; ++j)
      v[j] = x[((i64)bg * 64 + k) * 1024 + threadIdx.x + j * 256];
#pragma unroll
    for (int m = 0; m < 16; ++m) {
      float cv = cs[k * 16 + m];
#pragma unroll
      for (int j = 0; j < 4; ++j) acc[m][j] += cv * v[j];
    }
  }
#pragma unroll
  for (int m = 0; m < 16; ++m) {
    i64 base = (i64)(m >> 1) * (1024 * 1024) + ((i64)bg * 2 + (m & 1)) * 1024;
#pragma unroll
    for (int j = 0; j < 4; ++j)
      out[base + threadIdx.x + j * 256] = (bf16)acc[m][j];
  }
}

// seq[bg][m][c] = out1_by_e[e][(bg*2+s)][c] * expert_gates[e] * gates[bg]
__global__ __launch_bounds__(256) void k_seqscale(const float* __restrict__ out1,
                                                  const float* __restrict__ gates,
                                                  const float* __restrict__ eg,
                                                  float* __restrict__ seq, i64 total) {
  i64 idx = (i64)blockIdx.x * blockDim.x + threadIdx.x;
  if (idx >= total) return;
  int c = (int)(idx & 1023);
  i64 t = idx >> 10;
  int m = (int)(t & 15);
  i64 bg = t >> 4;
  int e = m >> 1, s = m & 1;
  float v = out1[((i64)e * 1024 + bg * 2 + s) * 1024 + c] * eg[e] * gates[bg];
  seq[idx] = v;
}

// final[bg][k][c] = sum_m combine[bg][k][m] * out2row[bg*16+m][c]
__global__ __launch_bounds__(256) void k_final(const float* __restrict__ comb,
                                               const float* __restrict__ out2,
                                               float* __restrict__ out) {
  int bg = blockIdx.x;
  __shared__ float cs[64 * 16];
  for (int i = threadIdx.x; i < 1024; i += 256) cs[i] = comb[(i64)bg * 1024 + i];
  float o2r[16][4];
  const float* o2 = out2 + (i64)bg * 16 * 1024;  // (b,2048,1024): row b*2048+g*16 = bg*16
#pragma unroll
  for (int m = 0; m < 16; ++m)
#pragma unroll
    for (int j = 0; j < 4; ++j)
      o2r[m][j] = o2[(i64)m * 1024 + threadIdx.x + j * 256];
  __syncthreads();
  for (int k = 0; k < 64; ++k) {
    float r[4] = {0.f, 0.f, 0.f, 0.f};
#pragma unroll
    for (int m = 0; m < 16; ++m) {
      float cv = cs[k * 16 + m];
#pragma unroll
      for (int j = 0; j < 4; ++j) r[j] += cv * o2r[m][j];
    }
#pragma unroll
    for (int j = 0; j < 4; ++j)
      out[((i64)bg * 64 + k) * 1024 + threadIdx.x + j * 256] = r[j];
  }
}

// ---------------------------------------------------------------------------
extern "C" void kernel_launch(void* const* d_in, const int* in_sizes, int n_in,
                              void* d_out, int out_size, void* d_ws, size_t ws_size,
                              hipStream_t stream) {
  (void)in_sizes; (void)n_in; (void)out_size; (void)ws_size;
  const float* x        = (const float*)d_in[0];
  const float* gamma1   = (const float*)d_in[1];
  const float* W_slot1  = (const float*)d_in[2];
  const float* gslot1   = (const float*)d_in[3];
  const float* W1f      = (const float*)d_in[4];
  const float* b1f      = (const float*)d_in[5];
  const float* W2f      = (const float*)d_in[6];
  const float* b2f      = (const float*)d_in[7];
  const float* eg       = (const float*)d_in[8];
  const float* Wg       = (const float*)d_in[9];
  const float* bg0      = (const float*)d_in[10];
  const float* gamma2   = (const float*)d_in[11];
  const float* W_slot2  = (const float*)d_in[12];
  const float* gslot2   = (const float*)d_in[13];
  const float* W1s      = (const float*)d_in[14];
  const float* b1s      = (const float*)d_in[15];
  const float* W2s      = (const float*)d_in[16];
  const float* b2s      = (const float*)d_in[17];
  float* out = (float*)d_out;

  // ---- workspace arena (lifetime-disjoint slab reuse), ~404 MB ----
  char* ws = (char*)d_ws;
  const size_t MB = 1ull << 20;
  // transposed bf16 weights (persist whole call), layout (N,K)
  bf16* wslot1_t = (bf16*)(ws + 0);            // 32MB (16384,1024)
  bf16* w1f_t    = (bf16*)(ws + 32 * MB);      // 64MB 8x(4096,1024)
  bf16* w2f_t    = (bf16*)(ws + 96 * MB);      // 64MB 8x(1024,4096)
  bf16* wslot2_t = (bf16*)(ws + 160 * MB);     // 8MB  (4096,1024)
  bf16* w1s_t    = (bf16*)(ws + 168 * MB);     // 32MB 4x(4096,1024)
  bf16* w2s_t    = (bf16*)(ws + 200 * MB);     // 32MB 4x(1024,4096)
  char* slabA = ws + 232 * MB;  // 64MB: xn_b | h_b | logits2_f | h2_b
  char* slabB = ws + 296 * MB;  // 32MB: slot_pre | out1_f | xs_f | comb2_b
  char* slabC = ws + 328 * MB;  // 32MB: seq_f | se2pre_f | dispT_b | out2_f
  char* slabD = ws + 360 * MB;  // 16MB: se_b | slots_b | xs_b | o2_b
  char* slabE = ws + 376 * MB;  // 16MB: se2_b | slots2_b
  float* gm_f     = (float*)(ws + 392 * MB);   // 2MB
  bf16*  gm_b     = (bf16*) (ws + 394 * MB);   // 1MB
  float* logits_f = (float*)(ws + 395 * MB);   // 2MB
  float* comb_f   = (float*)(ws + 397 * MB);   // 2MB (lives until k_final)
  bf16*  cmean_b  = (bf16*) (ws + 399 * MB);   // 4MB
  float* gates_f  = (float*)(ws + 403 * MB);   // 2KB

  bf16* xn_b     = (bf16*)slabA;  // ph2-4
  bf16* h_b      = (bf16*)slabA;  // ph6-7
  float* logits2 = (float*)slabA; // ph12-13
  bf16* h2_b     = (bf16*)slabA;  // ph15-16
  float* slotpre = (float*)slabB; // ph3
  float* out1_f  = (float*)slabB; // ph7-8
  float* xs_f    = (float*)slabB; // ph9-10
  bf16* comb2_b  = (bf16*)slabB;  // ph13-17
  float* seq_f   = (float*)slabC; // ph8-9
  float* se2pre  = (float*)slabC; // ph11
  bf16* dispT_b  = (bf16*)slabC;  // ph13-14
  float* out2_f  = (float*)slabC; // ph17-18
  bf16* se_b     = (bf16*)slabD;  // ph3-4
  bf16* slots_b  = (bf16*)slabD;  // ph5-6
  bf16* xs_b     = (bf16*)slabD;  // ph9-14
  bf16* o2_b     = (bf16*)slabD;  // ph16-17
  bf16* se2_b    = (bf16*)slabE;  // ph11-12
  bf16* slots2_b = (bf16*)slabE;  // ph14-15

  // ---- phase 1: weights fp32 -> bf16, transposed to (N,K) ----
  k_cvt_t_bf16<<<dim3(512, 32, 1), 256, 0, stream>>>(W_slot1, wslot1_t, 1024, 16384);
  k_cvt_t_bf16<<<dim3(128, 32, 8), 256, 0, stream>>>(W1f, w1f_t, 1024, 4096);
  k_cvt_t_bf16<<<dim3(32, 128, 8), 256, 0, stream>>>(W2f, w2f_t, 4096, 1024);
  k_cvt_t_bf16<<<dim3(128, 32, 1), 256, 0, stream>>>(W_slot2, wslot2_t, 1024, 4096);
  k_cvt_t_bf16<<<dim3(128, 32, 4), 256, 0, stream>>>(W1s, w1s_t, 1024, 4096);
  k_cvt_t_bf16<<<dim3(32, 128, 4), 256, 0, stream>>>(W2s, w2s_t, 4096, 1024);

  // ---- phase 2: xn = rmsnorm(x), gm = group mean ----
  k_rmsnorm<<<32768, 256, 0, stream>>>(x, gamma1, nullptr, xn_b, 1024, 0, 0, 0);
  k_group_mean<<<2048, 256, 0, stream>>>(x, gm_f, gm_b, 1024, 64, 524288LL);

  // ---- phase 3: slot_pre = gm @ W_slot1 ; se = rmsnorm ----
  k_gemm<1><<<dim3(128, 4, 1), 256, 0, stream>>>(
      gm_b, 1024, 0, wslot1_t, 1024, 0, 1,
      slotpre, nullptr, 16384, 0, nullptr, 0, 1024, 0);
  k_rmsnorm<<<8192, 256, 0, stream>>>(slotpre, gslot1, nullptr, se_b, 1024, 0, 0, 0);

  // ---- phase 4: logits = xn . se^T (per group), combine = softmax ----
  k_gemm16<<<dim3(4, 1, 512), 32, 0, stream>>>(
      xn_b, 1024, 65536, se_b, 1024, 16384, logits_f, 16, 1024, 1024);
  k_softmax_row<<<32768, 256, 0, stream>>>(logits_f, comb_f, nullptr, 16);

  // ---- phase 5: gates, slots (per-expert bf16 layout) ----
  k_gates<<<512, 256, 0, stream>>>(gm_f, Wg, bg0, gates_f, 1024);
  k_slots<<<512, 256, 0, stream>>>(x, comb_f, slots_b);

  // ---- phase 6/7: first-stage expert FFN ----
  k_gemm<1><<<dim3(32, 8, 8), 256, 0, stream>>>(          // h = gelu(slots@W1f+b1f)
      slots_b, 1024, 1048576, w1f_t, 1024, 4194304, 8,
      nullptr, h_b, 4096, 4194304, b1f, 4096, 1024, 1);
  k_gemm<1><<<dim3(8, 8, 8), 256, 0, stream>>>(           // out1 = h@W2f+b2f
      h_b, 4096, 4194304, w2f_t, 4096, 4194304, 8,
      out1_f, nullptr, 1024, 1048576, b2f, 1024, 4096, 0);

  // ---- phase 8/9: seq = out1 * eg * gates ; xs = rmsnorm(seq) ----
  k_seqscale<<<32768, 256, 0, stream>>>(out1_f, gates_f, eg, seq_f, 8388608LL);
  k_rmsnorm<<<8192, 256, 0, stream>>>(seq_f, gamma2, xs_f, xs_b, 1024, 0, 0, 0);

  // ---- phase 10/11: cmean, se2 ----
  k_group_mean<<<8192, 256, 0, stream>>>(xs_f, nullptr, cmean_b, 1024, 4, 2097152LL);
  k_gemm<1><<<dim3(32, 16, 1), 256, 0, stream>>>(
      cmean_b, 1024, 0, wslot2_t, 1024, 0, 1,
      se2pre, nullptr, 4096, 0, nullptr, 0, 1024, 0);
  k_rmsnorm<<<8192, 256, 0, stream>>>(se2pre, gslot2, nullptr, se2_b, 1024, 1, 512, 4);

  // ---- phase 12/13: logits2 = xs . se2^T ; comb2 / dispatchT softmaxes ----
  k_gemm<1><<<dim3(16, 16, 4), 256, 0, stream>>>(
      xs_b, 1024, 2097152, se2_b, 1024, 2097152, 0,
      logits2, nullptr, 2048, 4194304, nullptr, 0, 1024, 0);
  k_softmax_row<<<8192, 256, 0, stream>>>(logits2, nullptr, comb2_b, 2048);
  k_colsoftmax_T<<<dim3(8, 4), 256, 0, stream>>>(logits2, dispT_b, 2048, 2048);

  // ---- phase 14: slots2 = dispatchT @ xs ----
  k_gemm<0><<<dim3(8, 16, 4), 256, 0, stream>>>(
      dispT_b, 2048, 4194304, xs_b, 1024, 2097152, 0,
      nullptr, slots2_b, 1024, 2097152, nullptr, 0, 2048, 0);

  // ---- phase 15/16: second-stage expert FFN (z = b*4+e) ----
  k_gemm<1><<<dim3(32, 4, 16), 256, 0, stream>>>(
      slots2_b, 1024, 524288, w1s_t, 1024, 4194304, 4,
      nullptr, h2_b, 4096, 2097152, b1s, 4096, 1024, 1);
  k_gemm<1><<<dim3(8, 4, 16), 256, 0, stream>>>(
      h2_b, 4096, 2097152, w2s_t, 4096, 4194304, 4,
      nullptr, o2_b, 1024, 524288, b2s, 1024, 4096, 0);

  // ---- phase 17: out2 = comb2 @ o2 ----
  k_gemm<0><<<dim3(8, 16, 4), 256, 0, stream>>>(
      comb2_b, 2048, 4194304, o2_b, 1024, 2097152, 0,
      out2_f, nullptr, 1024, 2097152, nullptr, 0, 2048, 0);

  // ---- phase 18: final recombine -> d_out ----
  k_final<<<512, 256, 0, stream>>>(comb_f, out2_f, out);
}
